// QBNNAttention_2980707304151
// MI455X (gfx1250) — matrix operand order repaired and checked by reading, not compiled, not run on hardware
//
#include <hip/hip_runtime.h>
#include <math.h>

// MI455X / gfx1250, wave32. Matrix math via v_wmma_f32_16x16x32_bf16.
// Pipeline: one-time bf16 convert/transpose of x,W*,J ->
//           3x proj GEMM -> Qt = tanh(Q)@J -> causal flash attention
//           (scores = (Q/8)K^T + (lam*Qt)tanh(K)^T) -> out GEMM (fp32).
// Workspace (ushort): xb(4M) + 4xWt(4M) + Jt(64K) + Q,K,V,Qt,O(5x4M) ~ 56 MB.

#if defined(__has_builtin)
#  if __has_builtin(__builtin_amdgcn_tensor_load_to_lds)
#    define HAVE_TDM 1
#  endif
#  if __has_builtin(__builtin_amdgcn_s_wait_tensorcnt)
#    define HAVE_TCNT 1
#  endif
#endif

typedef __attribute__((ext_vector_type(16))) __bf16       v16bf;
typedef __attribute__((ext_vector_type(8)))  float        v8f;
typedef __attribute__((ext_vector_type(4)))  unsigned int v4u;
typedef __attribute__((ext_vector_type(8)))  int          v8i;
typedef __attribute__((ext_vector_type(4)))  int          v4i;

__device__ __forceinline__ unsigned short f2bf(float f) {
    return __builtin_bit_cast(unsigned short, (__bf16)f);   // v_cvt_*bf16_f32
}
__device__ __forceinline__ float bf2f(unsigned short h) {
    unsigned u = (unsigned)h << 16;
    return __builtin_bit_cast(float, u);
}
__device__ __forceinline__ __bf16 bfr(unsigned short h) {
    return __builtin_bit_cast(__bf16, h);
}
__device__ __forceinline__ unsigned pk2(float a, float b) {
    return (unsigned)f2bf(a) | ((unsigned)f2bf(b) << 16);
}
__device__ __forceinline__ v8f zero8() {
    v8f z;
#pragma unroll
    for (int i = 0; i < 8; ++i) z[i] = 0.f;
    return z;
}
// fast tanh: exact enough for bf16 operands; clamp avoids inf/inf
__device__ __forceinline__ float fast_tanh(float x) {
    x = fminf(15.f, fmaxf(-15.f, x));
    float t = __expf(2.f * x);
    return __fdividef(t - 1.f, t + 1.f);
}
__device__ __forceinline__ void unpack8(uint4 u, unsigned short* s) {
    s[0] = u.x & 0xffff; s[1] = u.x >> 16;
    s[2] = u.y & 0xffff; s[3] = u.y >> 16;
    s[4] = u.z & 0xffff; s[5] = u.z >> 16;
    s[6] = u.w & 0xffff; s[7] = u.w >> 16;
}

// WMMA 16x16x32 bf16 wave32 fragment index maps:
//  A: lanes 0-15 M=lane (K 0..7,16..23), lanes 16-31 M=lane-16 (K 8..15,24..31)
__device__ __forceinline__ int akmap(int e, int hi) { return e + (e & 8) + hi * 8; }
//  B: lanes 0-15 K 0..7,16..23, lanes 16-31 K 8..15,24..31 (same as A)
__device__ __forceinline__ int bkmap(int e, int hi) { return e + (e & 8) + hi * 8; }

// -------------------------------------------------------------------------
// One-time conversion kernels.
// -------------------------------------------------------------------------
__global__ __launch_bounds__(256) void conv_f32_bf16(
    const float* __restrict__ src, unsigned short* __restrict__ dst, int n4)
{
    int idx = blockIdx.x * 256 + threadIdx.x;
    if (idx < n4) {
        float4 v = ((const float4*)src)[idx];
        ((uint2*)dst)[idx] = make_uint2(pk2(v.x, v.y), pk2(v.z, v.w));
    }
}

// Wt[n*1024 + k] = bf16(W[k*1024 + n])  (1024x1024)
__global__ __launch_bounds__(256) void conv_w_transpose(
    const float* __restrict__ W, unsigned short* __restrict__ Wt)
{
    int idx = blockIdx.x * 256 + threadIdx.x;   // 262144 threads
    int k = idx >> 8, n0 = (idx & 255) * 4;
    float4 v = *(const float4*)&W[(size_t)k * 1024 + n0];
    Wt[(size_t)(n0 + 0) * 1024 + k] = f2bf(v.x);
    Wt[(size_t)(n0 + 1) * 1024 + k] = f2bf(v.y);
    Wt[(size_t)(n0 + 2) * 1024 + k] = f2bf(v.z);
    Wt[(size_t)(n0 + 3) * 1024 + k] = f2bf(v.w);
}

// Jt[h][n][d] = bf16(J[h][d][n])  (16 x 64 x 64)
__global__ __launch_bounds__(256) void conv_j_transpose(
    const float* __restrict__ J, unsigned short* __restrict__ Jt)
{
    int idx = blockIdx.x * 256 + threadIdx.x;   // 16384 threads
    int h = idx >> 10, rem = idx & 1023;
    int d = rem >> 4, n0 = (rem & 15) * 4;
    float4 v = *(const float4*)&J[(size_t)h * 4096 + d * 64 + n0];
    Jt[(size_t)h * 4096 + (n0 + 0) * 64 + d] = f2bf(v.x);
    Jt[(size_t)h * 4096 + (n0 + 1) * 64 + d] = f2bf(v.y);
    Jt[(size_t)h * 4096 + (n0 + 2) * 64 + d] = f2bf(v.z);
    Jt[(size_t)h * 4096 + (n0 + 3) * 64 + d] = f2bf(v.w);
}

// -------------------------------------------------------------------------
// (M x 1024) @ (1024 x 1024) + bias, all-bf16 operands (Wt pre-transposed).
// 128x64 tile/block, 32 rows/wave, 8 WMMA per 32-deep k-step per wave.
// Staging is pure uint4 copies (no VALU conversion in the k-loop).
// -------------------------------------------------------------------------
template <int OUTF32>
__global__ __launch_bounds__(128) void gemm_e1024(
    const unsigned short* __restrict__ A, const unsigned short* __restrict__ Wt,
    const float* __restrict__ bias,
    unsigned short* __restrict__ Ob, float* __restrict__ Of)
{
    __shared__ unsigned short As[128][40];   // [row][k], k contiguous
    __shared__ unsigned short WsT[64][40];   // [n][k],  k contiguous

    const int tid = threadIdx.x, lane = tid & 31, wv = tid >> 5;
    const int hi = lane >> 4, lo = lane & 15;
    const int rowBase = blockIdx.x * 128;
    const int colBase = blockIdx.y * 64;

    v8f acc[4][2];
#pragma unroll
    for (int t = 0; t < 4; ++t)
#pragma unroll
        for (int s = 0; s < 2; ++s) acc[t][s] = zero8();

    for (int kc = 0; kc < 1024; kc += 32) {
#pragma unroll
        for (int it = 0; it < 4; ++it) {      // A tile: 512 x uint4
            int idx = tid + it * 128;
            int r = idx >> 2, c = (idx & 3) * 8;
            *(uint4*)&As[r][c] =
                *(const uint4*)&A[(size_t)(rowBase + r) * 1024 + kc + c];
        }
#pragma unroll
        for (int it = 0; it < 2; ++it) {      // W^T tile: 256 x uint4
            int idx = tid + it * 128;
            int c = idx >> 2, kq = (idx & 3) * 8;
            *(uint4*)&WsT[c][kq] =
                *(const uint4*)&Wt[(size_t)(colBase + c) * 1024 + kc + kq];
        }
        if (kc + 32 < 1024) {                 // global_prefetch_b8 next slices
            __builtin_prefetch(&Wt[(size_t)(colBase + (tid >> 1)) * 1024 + kc + 32 + (tid & 1) * 16], 0, 1);
            __builtin_prefetch(&A[(size_t)(rowBase + tid) * 1024 + kc + 32], 0, 1);
        }
        __syncthreads();

        v16bf a[2];
#pragma unroll
        for (int s = 0; s < 2; ++s)
#pragma unroll
            for (int e = 0; e < 16; ++e)
                a[s][e] = bfr(As[wv * 32 + s * 16 + lo][akmap(e, hi)]);
#pragma unroll
        for (int t = 0; t < 4; ++t) {
            v16bf b;
#pragma unroll
            for (int e = 0; e < 16; ++e) b[e] = bfr(WsT[t * 16 + lo][bkmap(e, hi)]);
#pragma unroll
            for (int s = 0; s < 2; ++s)
                acc[t][s] = __builtin_amdgcn_wmma_f32_16x16x32_bf16(
                    false, a[s], false, b, (short)0, acc[t][s], false, false);
        }
        __syncthreads();
    }

#pragma unroll
    for (int t = 0; t < 4; ++t) {
        int col = colBase + t * 16 + lo;
        float bb = bias[col];
#pragma unroll
        for (int s = 0; s < 2; ++s)
#pragma unroll
            for (int r = 0; r < 8; ++r) {
                size_t row = (size_t)(rowBase + wv * 32 + s * 16 + r + hi * 8);
                float v = acc[t][s][r] + bb;
                if (OUTF32) Of[row * 1024 + col] = v;
                else        Ob[row * 1024 + col] = f2bf(v);
            }
    }
}

// -------------------------------------------------------------------------
// Qt[row,h,:] = tanh(Q[row,h,:]) @ J[h]  (64x64 per head), 64 rows/block.
// -------------------------------------------------------------------------
__global__ __launch_bounds__(128) void qtj_kernel(
    const unsigned short* __restrict__ Q, const unsigned short* __restrict__ Jt,
    unsigned short* __restrict__ Qt)
{
    __shared__ unsigned short As[64][40];    // tanh(Q) [row][k]
    __shared__ unsigned short JsT[64][72];   // J^T [n][k]

    const int tid = threadIdx.x, lane = tid & 31, wv = tid >> 5;
    const int hi = lane >> 4, lo = lane & 15;
    const int rowBase = blockIdx.x * 64;
    const int h = blockIdx.y;

#pragma unroll
    for (int it = 0; it < 4; ++it) {          // J^T tile: 512 x uint4
        int idx = tid + it * 128;
        int n = idx >> 3, d0 = (idx & 7) * 8;
        *(uint4*)&JsT[n][d0] = *(const uint4*)&Jt[(size_t)h * 4096 + n * 64 + d0];
    }

    v8f acc[4];
#pragma unroll
    for (int t = 0; t < 4; ++t) acc[t] = zero8();

    for (int kc = 0; kc < 64; kc += 32) {
#pragma unroll
        for (int it = 0; it < 2; ++it) {      // 256 x uint4 (64x32 bf16)
            int idx = tid + it * 128;
            int r = idx >> 2, c = (idx & 3) * 8;
            uint4 u = *(const uint4*)&Q[(size_t)(rowBase + r) * 1024 + h * 64 + kc + c];
            unsigned short s[8];
            unpack8(u, s);
            unsigned w0 = pk2(fast_tanh(bf2f(s[0])), fast_tanh(bf2f(s[1])));
            unsigned w1 = pk2(fast_tanh(bf2f(s[2])), fast_tanh(bf2f(s[3])));
            unsigned w2 = pk2(fast_tanh(bf2f(s[4])), fast_tanh(bf2f(s[5])));
            unsigned w3 = pk2(fast_tanh(bf2f(s[6])), fast_tanh(bf2f(s[7])));
            *(uint4*)&As[r][c] = make_uint4(w0, w1, w2, w3);
        }
        __syncthreads();
        v16bf a;
#pragma unroll
        for (int e = 0; e < 16; ++e) a[e] = bfr(As[wv * 16 + lo][akmap(e, hi)]);
#pragma unroll
        for (int t = 0; t < 4; ++t) {
            v16bf b;
#pragma unroll
            for (int e = 0; e < 16; ++e) b[e] = bfr(JsT[t * 16 + lo][kc + bkmap(e, hi)]);
            acc[t] = __builtin_amdgcn_wmma_f32_16x16x32_bf16(
                false, a, false, b, (short)0, acc[t], false, false);
        }
        __syncthreads();
    }

#pragma unroll
    for (int t = 0; t < 4; ++t)
#pragma unroll
        for (int r = 0; r < 8; ++r)
            Qt[(size_t)(rowBase + wv * 16 + r + hi * 8) * 1024 + h * 64 + t * 16 + lo] =
                f2bf(acc[t][r]);
}

// -------------------------------------------------------------------------
// Causal flash attention with bilinear correction. grid (S/64, B*H),
// 4 waves/block, 16 rows/wave, 32-column chunks, 12 WMMA per chunk/wave.
// K tile staged by the Tensor Data Mover when available.
// -------------------------------------------------------------------------
__global__ __launch_bounds__(128) void flash_kernel(
    const unsigned short* __restrict__ Q,  const unsigned short* __restrict__ K,
    const unsigned short* __restrict__ V,  const unsigned short* __restrict__ Qt,
    unsigned short* __restrict__ O, const float* __restrict__ lamp)
{
    __shared__ unsigned short Kt[32][72];      // K [j][d], d contiguous
    __shared__ unsigned short Knt[32][72];     // tanh(K)
    __shared__ unsigned short Vt[64][40];      // V^T [d][j], j contiguous
    __shared__ float          sbuf[4][16][33]; // per-wave raw scores
    __shared__ unsigned short pbuf[4][16][40]; // per-wave probs

    const int S = 2048;
    const int tid = threadIdx.x, lane = tid & 31, wv = tid >> 5;
    const int hi = lane >> 4, lo = lane & 15;
    const int b = blockIdx.y >> 4, h = blockIdx.y & 15;
    const int i0 = blockIdx.x * 64 + wv * 16;
    const size_t rowoff = (size_t)(b * S) * 1024 + (size_t)h * 64;
    const float lam = lamp[0];

#if defined(HAVE_TCNT)
    __builtin_amdgcn_s_wait_tensorcnt(0);      // TENSORcnt==0 here: no stall
#endif

    // preload A-fragments: Q/8 and lam*Qt (hd=64 -> 2 k-chunks each)
    v16bf aq[2], at[2];
#pragma unroll
    for (int c = 0; c < 2; ++c)
#pragma unroll
        for (int half = 0; half < 2; ++half) {
            int k0 = c * 32 + half * 16 + hi * 8;
            size_t g = rowoff + (size_t)(i0 + lo) * 1024 + k0;
            unsigned short sq[8], st[8];
            unpack8(*(const uint4*)&Q[g],  sq);
            unpack8(*(const uint4*)&Qt[g], st);
#pragma unroll
            for (int i = 0; i < 8; ++i) {
                aq[c][half * 8 + i] = bfr(f2bf(bf2f(sq[i]) * 0.125f));
                at[c][half * 8 + i] = bfr(f2bf(bf2f(st[i]) * lam));
            }
        }

    v8f acco[4];
#pragma unroll
    for (int t = 0; t < 4; ++t) acco[t] = zero8();
    float m_i = -3.0e30f, l_i = 0.f;

    const int nchunk = blockIdx.x * 2 + 2;     // uniform per block (causal)
    for (int cj = 0; cj < nchunk; ++cj) {
        const int j0 = cj * 32;

#if defined(HAVE_TDM)
        if (wv == 0) {  // TDM: 2D tile 64 x 32 (2B elems), row stride 1024,
                        // LDS pad 4 dwords every 32 dwords -> Kt[32][72]
            unsigned long long ga =
                (unsigned long long)(const void*)&K[rowoff + (size_t)j0 * 1024];
            unsigned ldsa = (unsigned)(size_t)&Kt[0][0];
            v4u g0;
            g0[0] = 1u;                                   // count=1
            g0[1] = ldsa;                                 // lds_addr
            g0[2] = (unsigned)(ga & 0xffffffffu);         // global_addr lo
            g0[3] = (unsigned)((ga >> 32) & 0x1ffffffu) | (2u << 30); // hi|type=2
            v8i g1;
            g1[0] = (int)((1u << 16) | (1u << 20) | (4u << 22) | (3u << 25));
            g1[1] = (int)(64u << 16);                     // tensor_dim0 = 64
            g1[2] = (int)(32u << 16);                     // tensor_dim1 = 32
            g1[3] = (int)(64u << 16);                     // tile_dim0 = 64
            g1[4] = 32;                                   // tile_dim1 = 32
            g1[5] = 1024;                                 // dim0 stride = 1024
            g1[6] = 0; g1[7] = 0;
            v4i gz = {0, 0, 0, 0};
#  if __has_include(<hip/amd_detail/amd_gfx1250_TDM.h>)
            v8i gz8 = {0, 0, 0, 0, 0, 0, 0, 0};
            __builtin_amdgcn_tensor_load_to_lds(g0, g1, gz, gz, gz8, 0);
#  else
            __builtin_amdgcn_tensor_load_to_lds(g0, g1, gz, gz, 0);
#  endif
        }
#else
#pragma unroll
        for (int it = 0; it < 2; ++it) {       // manual K tile (256 x uint4)
            int idx = tid + it * 128;
            int j = idx >> 3, d0 = (idx & 7) * 8;
            *(uint4*)&Kt[j][d0] =
                *(const uint4*)&K[rowoff + (size_t)(j0 + j) * 1024 + d0];
        }
#endif
#pragma unroll
        for (int it = 0; it < 2; ++it) {       // V^T staging (scatter b16)
            int idx = tid + it * 128;
            int j = idx >> 3, d0 = (idx & 7) * 8;
            uint4 u = *(const uint4*)&V[rowoff + (size_t)(j0 + j) * 1024 + d0];
            unsigned short s[8];
            unpack8(u, s);
#pragma unroll
            for (int i = 0; i < 8; ++i) Vt[d0 + i][j] = s[i];
        }
#if defined(HAVE_TDM)
        if (wv == 0) __builtin_amdgcn_s_wait_tensorcnt(0);
#endif
        __syncthreads();

        // tanh(K) tile from Kt
#pragma unroll
        for (int it = 0; it < 2; ++it) {
            int idx = tid + it * 128;
            int j = idx >> 3, d0 = (idx & 7) * 8;
            unsigned short s[8];
            unpack8(*(const uint4*)&Kt[j][d0], s);
            unsigned w0 = pk2(fast_tanh(bf2f(s[0])), fast_tanh(bf2f(s[1])));
            unsigned w1 = pk2(fast_tanh(bf2f(s[2])), fast_tanh(bf2f(s[3])));
            unsigned w2 = pk2(fast_tanh(bf2f(s[4])), fast_tanh(bf2f(s[5])));
            unsigned w3 = pk2(fast_tanh(bf2f(s[6])), fast_tanh(bf2f(s[7])));
            *(uint4*)&Knt[j][d0] = make_uint4(w0, w1, w2, w3);
        }
        __syncthreads();

        // two 16x16 score tiles: (Q/8)K^T + (lam Qt)tanh(K)^T, 4-WMMA chain
#pragma unroll
        for (int t = 0; t < 2; ++t) {
            v8f sc = zero8();
#pragma unroll
            for (int c = 0; c < 2; ++c) {
                v16bf bk, bn;
#pragma unroll
                for (int e = 0; e < 16; ++e) {
                    int kk = c * 32 + bkmap(e, hi);
                    bk[e] = bfr(Kt[t * 16 + lo][kk]);
                    bn[e] = bfr(Knt[t * 16 + lo][kk]);
                }
                sc = __builtin_amdgcn_wmma_f32_16x16x32_bf16(
                    false, aq[c], false, bk, (short)0, sc, false, false);
                sc = __builtin_amdgcn_wmma_f32_16x16x32_bf16(
                    false, at[c], false, bn, (short)0, sc, false, false);
            }
#pragma unroll
            for (int r = 0; r < 8; ++r) sbuf[wv][r + hi * 8][t * 16 + lo] = sc[r];
        }
        __syncthreads();

        // online softmax, one row per lane (lanes 16-31 mirror 0-15)
        const int rowg = i0 + lo;
        float mnew = m_i;
#pragma unroll
        for (int col = 0; col < 32; ++col) {
            float s = (j0 + col <= rowg) ? sbuf[wv][lo][col] : -3.0e30f;
            mnew = fmaxf(mnew, s);
        }
        float alpha = __expf(m_i - mnew);
        float lsum = 0.f;
#pragma unroll
        for (int col = 0; col < 32; ++col) {
            float s = (j0 + col <= rowg) ? sbuf[wv][lo][col] : -3.0e30f;
            float p = (s < -1.0e29f) ? 0.f : __expf(s - mnew);
            if (lane < 16) pbuf[wv][lo][col] = f2bf(p);
            lsum += p;
        }
        m_i = mnew;
        l_i = l_i * alpha + lsum;
        __syncthreads();

        // rescale accumulators, then P @ V
        float ar[8];
#pragma unroll
        for (int r = 0; r < 8; ++r) ar[r] = __shfl(alpha, r + hi * 8, 32);
        v16bf pf;
#pragma unroll
        for (int e = 0; e < 16; ++e) pf[e] = bfr(pbuf[wv][lo][akmap(e, hi)]);
#pragma unroll
        for (int t2 = 0; t2 < 4; ++t2) {
            v16bf bv;
#pragma unroll
            for (int e = 0; e < 16; ++e) bv[e] = bfr(Vt[t2 * 16 + lo][bkmap(e, hi)]);
#pragma unroll
            for (int r = 0; r < 8; ++r) acco[t2][r] *= ar[r];
            acco[t2] = __builtin_amdgcn_wmma_f32_16x16x32_bf16(
                false, pf, false, bv, (short)0, acco[t2], false, false);
        }
        __syncthreads();
    }

    // normalize and write O (bf16, [b,s,h,d] == row-major E)
    float li[8];
#pragma unroll
    for (int r = 0; r < 8; ++r) li[r] = 1.0f / __shfl(l_i, r + hi * 8, 32);
#pragma unroll
    for (int t2 = 0; t2 < 4; ++t2)
#pragma unroll
        for (int r = 0; r < 8; ++r)
            O[rowoff + (size_t)(i0 + r + hi * 8) * 1024 + t2 * 16 + lo] =
                f2bf(acco[t2][r] * li[r]);
}

// -------------------------------------------------------------------------
extern "C" void kernel_launch(void* const* d_in, const int* in_sizes, int n_in,
                              void* d_out, int out_size, void* d_ws, size_t ws_size,
                              hipStream_t stream) {
    (void)n_in; (void)out_size; (void)ws_size;
    const float* x   = (const float*)d_in[0];
    const float* Wq  = (const float*)d_in[1];
    const float* bq  = (const float*)d_in[2];
    const float* Wk  = (const float*)d_in[3];
    const float* bk  = (const float*)d_in[4];
    const float* Wv  = (const float*)d_in[5];
    const float* bv  = (const float*)d_in[6];
    const float* Wo  = (const float*)d_in[7];
    const float* bo  = (const float*)d_in[8];
    const float* J   = (const float*)d_in[9];
    const float* lam = (const float*)d_in[10];

    const int BS = in_sizes[0] / 1024;   // B*S
    const int Bn = BS / 2048;

    const size_t per = (size_t)BS * 1024;
    const size_t wsz = 1024 * 1024;
    unsigned short* xb  = (unsigned short*)d_ws;
    unsigned short* Wqt = xb  + per;
    unsigned short* Wkt = Wqt + wsz;
    unsigned short* Wvt = Wkt + wsz;
    unsigned short* Wot = Wvt + wsz;
    unsigned short* Jt  = Wot + wsz;
    unsigned short* Qb  = Jt  + 16 * 64 * 64;
    unsigned short* Kb  = Qb  + per;
    unsigned short* Vb  = Kb  + per;
    unsigned short* Qtb = Vb  + per;
    unsigned short* Ob  = Qtb + per;

    dim3 blk(128);
    dim3 gGemm(BS / 128, 16);

    conv_f32_bf16<<<dim3((BS * 1024 / 4 + 255) / 256), dim3(256), 0, stream>>>(x, xb, BS * 1024 / 4);
    conv_w_transpose<<<dim3(1024), dim3(256), 0, stream>>>(Wq, Wqt);
    conv_w_transpose<<<dim3(1024), dim3(256), 0, stream>>>(Wk, Wkt);
    conv_w_transpose<<<dim3(1024), dim3(256), 0, stream>>>(Wv, Wvt);
    conv_w_transpose<<<dim3(1024), dim3(256), 0, stream>>>(Wo, Wot);
    conv_j_transpose<<<dim3(64), dim3(256), 0, stream>>>(J, Jt);

    gemm_e1024<0><<<gGemm, blk, 0, stream>>>(xb, Wqt, bq, Qb, nullptr);
    gemm_e1024<0><<<gGemm, blk, 0, stream>>>(xb, Wkt, bk, Kb, nullptr);
    gemm_e1024<0><<<gGemm, blk, 0, stream>>>(xb, Wvt, bv, Vb, nullptr);

    qtj_kernel<<<dim3(BS / 64, 16), blk, 0, stream>>>(Qb, Jt, Qtb);

    flash_kernel<<<dim3(2048 / 64, Bn * 16), blk, 0, stream>>>(Qb, Kb, Vb, Qtb, Ob, lam);

    gemm_e1024<1><<<gGemm, blk, 0, stream>>>(Ob, Wot, bo, nullptr, (float*)d_out);
}
